// MultiLevelEncoder_49082886259265
// MI455X (gfx1250) — compile-verified, matrix operand order checked by reading, not run on hardware
//
#include <hip/hip_runtime.h>
#include <hip/hip_bf16.h>
#include <stdint.h>

typedef _Float16 half4 __attribute__((ext_vector_type(4)));
typedef _Float16 half8 __attribute__((ext_vector_type(8)));
typedef _Float16 v16h  __attribute__((ext_vector_type(16)));
typedef float    v8f   __attribute__((ext_vector_type(8)));
typedef unsigned int u32x4 __attribute__((ext_vector_type(4)));
typedef int          i32x8 __attribute__((ext_vector_type(8)));
typedef int          i32x4 __attribute__((ext_vector_type(4)));

#define TOK   12544   // B*S
#define BD    256
#define SD    49
#define DD    512
#define HD    8
#define DKD   64
#define DFD   2048
#define GRIDD 7
#define NLAY  3

#define CAT16(lo, hi) __builtin_shufflevector(lo, hi, 0,1,2,3,4,5,6,7,8,9,10,11,12,13,14,15)

#if __has_builtin(__builtin_amdgcn_tensor_load_to_lds) && __has_builtin(__builtin_amdgcn_s_wait_tensorcnt)
#define USE_TDM 1
#else
#define USE_TDM 0
#endif

// ---------------- elementwise f32 -> f16 ----------------
__global__ void k_f32_to_f16(const float* __restrict__ src, _Float16* __restrict__ dst, int n) {
  int i = blockIdx.x * blockDim.x + threadIdx.x;
  if (i < n) dst[i] = (_Float16)src[i];
}

// ---------------- weight transpose+convert: w[K,N] -> wt[N,K] f16 ----------------
__global__ void k_wt_transpose(const float* __restrict__ w, _Float16* __restrict__ wt, int K, int N) {
  int i = blockIdx.x * blockDim.x + threadIdx.x;
  if (i >= K * N) return;
  int k = i / N, n = i - k * N;
  wt[(size_t)n * K + k] = (_Float16)w[i];
}

// ---------------- padding mask: mask[b*S+s] = (sum_d x == 0) ----------------
__global__ __launch_bounds__(128)
void k_mask(const float* __restrict__ x, int* __restrict__ mask) {
  __shared__ float red[128];
  int row = blockIdx.x;
  float s = 0.f;
  for (int d = threadIdx.x; d < DD; d += 128) s += x[(size_t)row * DD + d];
  red[threadIdx.x] = s; __syncthreads();
  for (int st = 64; st > 0; st >>= 1) {
    if (threadIdx.x < st) red[threadIdx.x] += red[threadIdx.x + st];
    __syncthreads();
  }
  if (threadIdx.x == 0) mask[row] = (red[0] == 0.0f) ? 1 : 0;
}

// ---------------- geometric relative bias: relu(emb . wg_w + wg_b) -> [H,S,S] ----------------
__global__ void k_disrel(const float* __restrict__ wgw, const float* __restrict__ wgb,
                         float* __restrict__ out) {
  int i = blockIdx.x * blockDim.x + threadIdx.x;
  if (i >= HD * SD * SD) return;
  int h = i / (SD * SD);
  int r = i - h * SD * SD;
  int q = r / SD, k = r - q * SD;
  float cxq = (2.0f * (float)(q / GRIDD) + 1.0f) * (0.5f / GRIDD);
  float cyq = (2.0f * (float)(q % GRIDD) + 1.0f) * (0.5f / GRIDD);
  float cxk = (2.0f * (float)(k / GRIDD) + 1.0f) * (0.5f / GRIDD);
  float cyk = (2.0f * (float)(k % GRIDD) + 1.0f) * (0.5f / GRIDD);
  float tx = cxq - cxk, ty = cyq - cyk;
  float pm[4];
  pm[0] = logf(fmaxf(fabsf(tx * 0.5f), 0.001f));
  pm[1] = logf(fmaxf(fabsf(ty * 0.5f), 0.001f));
  pm[2] = logf(fmaxf(fabsf((tx + ty) * 0.5f), 0.001f));
  pm[3] = logf(fmaxf(fabsf(sqrtf(tx * tx + ty * ty) * 0.5f), 0.001f));
  float acc = wgb[h];
  const float lw = logf(1000.0f) / 8.0f;
  for (int j = 0; j < 4; ++j)
    for (int f = 0; f < 8; ++f) {
      float mul = 100.0f * pm[j] * expf(-lw * (float)f);
      int g = j * 8 + f;
      acc += sinf(mul) * wgw[h * 64 + g] + cosf(mul) * wgw[h * 64 + 32 + g];
    }
  out[i] = fmaxf(acc, 0.0f);
}

// ---------------- Swin relative position bias -> [S,S] ----------------
__global__ void k_posrel(const float* __restrict__ table, float* __restrict__ out) {
  int i = blockIdx.x * blockDim.x + threadIdx.x;
  if (i >= SD * SD) return;
  int q = i / SD, k = i - q * SD;
  int r0 = q / GRIDD - k / GRIDD + (GRIDD - 1);
  int r1 = q % GRIDD - k % GRIDD + (GRIDD - 1);
  out[i] = table[r0 * (2 * GRIDD - 1) + r1];
}

// ---------------- TDM: async DMA of a rows x 32-half tile into padded LDS ----------------
// LDS row stride = 32 halves + 24 pad halves = 56 (pad_interval=3 -> every 16 DWORDs,
// pad_amount=11 -> 12 DWORDs), matching the GEMM's conflict-free LDS layout.
#define LDG 56
#if USE_TDM
__device__ static inline void tdm_load_tile(const _Float16* gsrc, uint32_t lds_byte,
                                            int rows, int K) {
  unsigned long long ga = (unsigned long long)(uintptr_t)gsrc;
  u32x4 g0;
  g0[0] = 1u;                                    // count=1, user descriptor
  g0[1] = lds_byte;                              // LDS dest byte address
  g0[2] = (uint32_t)ga;                          // global_addr[31:0]
  g0[3] = (uint32_t)((ga >> 32) & 0x1FFFFFFu) | (2u << 30);  // addr[56:32] | type=2
  i32x8 g1;
  g1[0] = (1 << 16) | (1 << 20) | (3 << 22) | (11 << 25);  // data=2B, pad_en, int=16DW, amt=12DW
  g1[1] = 32 << 16;                              // tensor_dim0 = 32 (low 16 bits)
  g1[2] = (rows & 0xFFFF) << 16;                 // tensor_dim0 hi=0 | tensor_dim1 lo = rows
  g1[3] = 32 << 16;                              // tensor_dim1 hi=0 | tile_dim0 = 32
  g1[4] = rows & 0xFFFF;                         // tile_dim1 = rows, tile_dim2 = 0
  g1[5] = K;                                     // tensor_dim0_stride (row pitch, elements)
  g1[6] = 0;
  g1[7] = 0;
  i32x4 z4 = {0, 0, 0, 0};                       // groups 2/3 unused (2D tile)
  i32x8 z8 = {0, 0, 0, 0, 0, 0, 0, 0};
  __builtin_amdgcn_tensor_load_to_lds(g0, g1, z4, z4, z8, 0);  // 6-arg clang-23 form
}
#endif

// ---------------- WMMA GEMM: C = A[M,K] * BT[N,K]^T + bias, optional relu ----------------
// block 128x64, 512 threads = 16 waves, each wave -> 32x16 C strip (2 accumulators).
// TDM path: double-buffered LDS tiles filled by the Tensor Data Mover (wave 0 issues,
// TENSORcnt retires), compute waves only do fragment ds_loads + v_wmma.
__global__ __launch_bounds__(512)
void k_gemm(const _Float16* __restrict__ A, const _Float16* __restrict__ BT,
            const float* __restrict__ bias, float* __restrict__ Cf,
            _Float16* __restrict__ Ch, int M, int N, int K, int relu) {
#if USE_TDM
  __shared__ _Float16 As[2][128][LDG];
  __shared__ _Float16 Bs[2][64][LDG];
#else
  __shared__ _Float16 As[1][128][LDG];
  __shared__ _Float16 Bs[1][64][LDG];
#endif
  const int tid  = threadIdx.x;
  const int wave = tid >> 5, lane = tid & 31;
  const int bm = blockIdx.x * 128, bn = blockIdx.y * 64;
  const int wm = (wave >> 2) * 32, wn = (wave & 3) * 16;
  const int lm   = lane & 15;
  const int kb   = ((lane >> 4) & 1) * 8;            // ISA A/B fragment K split
  const int moff = (lane >> 4) << 3;                 // ISA C fragment M split

  v8f acc0 = {}; v8f acc1 = {};

#if USE_TDM
  const _Float16* Abase = A  + (size_t)bm * K;
  const _Float16* Bbase = BT + (size_t)bn * K;
  const uint32_t ldsA[2] = { (uint32_t)(uintptr_t)&As[0][0][0],
                             (uint32_t)(uintptr_t)&As[1][0][0] };
  const uint32_t ldsB[2] = { (uint32_t)(uintptr_t)&Bs[0][0][0],
                             (uint32_t)(uintptr_t)&Bs[1][0][0] };
  if (tid < 32) {                 // wave 0 drives the DMA
    tdm_load_tile(Abase, ldsA[0], 128, K);
    tdm_load_tile(Bbase, ldsB[0], 64, K);
  }
  int buf = 0;
  for (int k0 = 0; k0 < K; k0 += 32) {
    const int nbuf = buf ^ 1;
    if (tid < 32) {
      if (k0 + 32 < K) {          // prefetch next tiles into the other buffer
        tdm_load_tile(Abase + k0 + 32, ldsA[nbuf], 128, K);
        tdm_load_tile(Bbase + k0 + 32, ldsB[nbuf], 64, K);
        __builtin_amdgcn_s_wait_tensorcnt(2);   // current pair retired, next in flight
      } else {
        __builtin_amdgcn_s_wait_tensorcnt(0);
      }
    }
    __syncthreads();
    const _Float16 (*Ac)[LDG] = As[buf];
    const _Float16 (*Bc)[LDG] = Bs[buf];

    half8 b0 = *(const half8*)&Bc[wn + lm][kb];
    half8 b1 = *(const half8*)&Bc[wn + lm][kb + 16];
    v16h bf = CAT16(b0, b1);
    half8 a0 = *(const half8*)&Ac[wm + lm][kb];
    half8 a1 = *(const half8*)&Ac[wm + lm][kb + 16];
    v16h af0 = CAT16(a0, a1);
    half8 a2 = *(const half8*)&Ac[wm + 16 + lm][kb];
    half8 a3 = *(const half8*)&Ac[wm + 16 + lm][kb + 16];
    v16h af1 = CAT16(a2, a3);

    acc0 = __builtin_amdgcn_wmma_f32_16x16x32_f16(false, af0, false, bf, (short)0, acc0, false, false);
    acc1 = __builtin_amdgcn_wmma_f32_16x16x32_f16(false, af1, false, bf, (short)0, acc1, false, false);
    __syncthreads();
    buf = nbuf;
  }
#else
  const int arow = tid >> 2, acol = (tid & 3) * 8;   // A tile: 128x32, 8 halves/thread
  const int brow = tid >> 3, bcol = (tid & 7) * 4;   // B tile: 64x32, 4 halves/thread
  const _Float16* ag = A  + (size_t)(bm + arow) * K + acol;
  const _Float16* bg = BT + (size_t)(bn + brow) * K + bcol;
  for (int k0 = 0; k0 < K; k0 += 32) {
    __syncthreads();
    *(half8*)&As[0][arow][acol] = *(const half8*)(ag + k0);
    *(half4*)&Bs[0][brow][bcol] = *(const half4*)(bg + k0);
    if (k0 + 32 < K) {
      __builtin_prefetch(ag + k0 + 32, 0, 1);
      __builtin_prefetch(bg + k0 + 32, 0, 1);
    }
    __syncthreads();
    half8 b0 = *(const half8*)&Bs[0][wn + lm][kb];
    half8 b1 = *(const half8*)&Bs[0][wn + lm][kb + 16];
    v16h bf = CAT16(b0, b1);
    half8 a0 = *(const half8*)&As[0][wm + lm][kb];
    half8 a1 = *(const half8*)&As[0][wm + lm][kb + 16];
    v16h af0 = CAT16(a0, a1);
    half8 a2 = *(const half8*)&As[0][wm + 16 + lm][kb];
    half8 a3 = *(const half8*)&As[0][wm + 16 + lm][kb + 16];
    v16h af1 = CAT16(a2, a3);
    acc0 = __builtin_amdgcn_wmma_f32_16x16x32_f16(false, af0, false, bf, (short)0, acc0, false, false);
    acc1 = __builtin_amdgcn_wmma_f32_16x16x32_f16(false, af1, false, bf, (short)0, acc1, false, false);
  }
#endif

  const int col = bn + wn + lm;
  const float bv = bias ? bias[col] : 0.f;
  const int mbase = bm + wm + moff;
  for (int r = 0; r < 8; ++r) {
    float v0 = acc0[r] + bv;
    float v1 = acc1[r] + bv;
    if (relu) { v0 = fmaxf(v0, 0.f); v1 = fmaxf(v1, 0.f); }
    size_t i0 = (size_t)(mbase + r) * N + col;
    size_t i1 = (size_t)(mbase + 16 + r) * N + col;
    if (Cf) { Cf[i0] = v0; Cf[i1] = v1; }
    if (Ch) { Ch[i0] = (_Float16)v0; Ch[i1] = (_Float16)v1; }
  }
}

// ---------------- attention per (b,h), S=49 padded to 64, WMMA both matmuls ----------------
#define LDA 72  // 144B row stride: 16B aligned, conflict-free for 16-row fragments
__global__ __launch_bounds__(256)
void k_attn(const _Float16* __restrict__ Q, const _Float16* __restrict__ Km,
            const _Float16* __restrict__ V, const float* __restrict__ rel, int perHead,
            const int* __restrict__ mask, _Float16* __restrict__ O) {
  __shared__ _Float16 Qs[64][LDA];   // reused for P after softmax
  __shared__ _Float16 Ks[64][LDA];
  __shared__ _Float16 Vt[64][LDA];   // Vt[d][s] = V[s][d]
  __shared__ float    Sc[64][65];
  const int b = blockIdx.x >> 3;
  const int h = blockIdx.x & 7;
  const int tid = threadIdx.x;
  const int wave = tid >> 5, lane = tid & 31;
  const int lm = lane & 15;
  const int moff = (lane >> 4) << 3;

  // load Q,K,V tiles (rows >= S zero-padded)
  for (int idx = tid; idx < 64 * 8; idx += 256) {
    int row = idx >> 3, d = (idx & 7) * 8;
    half8 qv = {}, kv = {}, vv = {};
    if (row < SD) {
      size_t base = ((size_t)(b * SD + row)) * DD + h * DKD + d;
      qv = *(const half8*)(Q + base);
      kv = *(const half8*)(Km + base);
      vv = *(const half8*)(V + base);
    }
    *(half8*)&Qs[row][d] = qv;
    *(half8*)&Ks[row][d] = kv;
    for (int j = 0; j < 8; ++j) Vt[d + j][row] = vv[j];
  }
  __syncthreads();

  // scores = Q*K^T/8 + rel, masked
  for (int tile = wave; tile < 16; tile += 8) {
    int tm = (tile >> 2) * 16, tn = (tile & 3) * 16;
    v8f acc = {};
    for (int k0 = 0; k0 < 64; k0 += 32) {
      int kk0 = k0 + moff;
      half8 a0 = *(const half8*)&Qs[tm + lm][kk0];
      half8 a1 = *(const half8*)&Qs[tm + lm][kk0 + 16];
      v16h af = CAT16(a0, a1);
      half8 b0 = *(const half8*)&Ks[tn + lm][kk0];
      half8 b1 = *(const half8*)&Ks[tn + lm][kk0 + 16];
      v16h bf = CAT16(b0, b1);
      acc = __builtin_amdgcn_wmma_f32_16x16x32_f16(false, af, false, bf, (short)0, acc, false, false);
    }
    int kcol = tn + lm;
    int msk = (kcol >= SD) ? 1 : mask[b * SD + kcol];
    for (int r = 0; r < 8; ++r) {
      int qi = tm + moff + r;
      float val = acc[r] * 0.125f;
      if (qi < SD && kcol < SD)
        val += perHead ? rel[((size_t)h * SD + qi) * SD + kcol] : rel[qi * SD + kcol];
      if (msk) val = -1e30f;
      Sc[qi][kcol] = val;
    }
  }
  __syncthreads();

  // softmax rows -> P (f16, into Qs)
  if (tid < 64) {
    float mx = -1e30f;
    for (int k = 0; k < SD; ++k) mx = fmaxf(mx, Sc[tid][k]);
    float sum = 0.f;
    for (int k = 0; k < SD; ++k) sum += expf(Sc[tid][k] - mx);
    float inv = 1.0f / sum;
    for (int k = 0; k < 64; ++k) {
      float p = (k < SD) ? expf(Sc[tid][k] - mx) * inv : 0.f;
      Qs[tid][k] = (_Float16)p;
    }
  }
  __syncthreads();

  // O = P * V
  for (int tile = wave; tile < 16; tile += 8) {
    int tm = (tile >> 2) * 16, tn = (tile & 3) * 16;
    v8f acc = {};
    for (int k0 = 0; k0 < 64; k0 += 32) {
      int kk0 = k0 + moff;
      half8 a0 = *(const half8*)&Qs[tm + lm][kk0];
      half8 a1 = *(const half8*)&Qs[tm + lm][kk0 + 16];
      v16h af = CAT16(a0, a1);
      half8 b0 = *(const half8*)&Vt[tn + lm][kk0];
      half8 b1 = *(const half8*)&Vt[tn + lm][kk0 + 16];
      v16h bf = CAT16(b0, b1);
      acc = __builtin_amdgcn_wmma_f32_16x16x32_f16(false, af, false, bf, (short)0, acc, false, false);
    }
    int dcol = tn + lm;
    for (int r = 0; r < 8; ++r) {
      int qi = tm + moff + r;
      if (qi < SD)
        O[((size_t)(b * SD + qi)) * DD + h * DKD + dcol] = (_Float16)acc[r];
    }
  }
}

// ---------------- fused residual + layernorm: y = LN(x + c0*a + c1*b)*g + be ----------------
__global__ __launch_bounds__(256)
void k_res_ln(const float* __restrict__ x, const float* __restrict__ a,
              const float* __restrict__ bb, float c0, float c1,
              const float* __restrict__ g, const float* __restrict__ be,
              float* __restrict__ yf, _Float16* __restrict__ yh) {
  __shared__ float red[256];
  const int row = blockIdx.x;
  const int t = threadIdx.x;
  size_t base = (size_t)row * DD;
  float t0 = x[base + t], t1 = x[base + t + 256];
  if (a)  { t0 += c0 * a[base + t];  t1 += c0 * a[base + t + 256]; }
  if (bb) { t0 += c1 * bb[base + t]; t1 += c1 * bb[base + t + 256]; }
  red[t] = t0 + t1; __syncthreads();
  for (int s = 128; s > 0; s >>= 1) { if (t < s) red[t] += red[t + s]; __syncthreads(); }
  float mean = red[0] * (1.0f / DD);
  __syncthreads();
  float d0 = t0 - mean, d1 = t1 - mean;
  red[t] = d0 * d0 + d1 * d1; __syncthreads();
  for (int s = 128; s > 0; s >>= 1) { if (t < s) red[t] += red[t + s]; __syncthreads(); }
  float inv = rsqrtf(red[0] * (1.0f / DD) + 1e-5f);
  float y0 = d0 * inv * g[t] + be[t];
  float y1 = d1 * inv * g[t + 256] + be[t + 256];
  if (yf) { yf[base + t] = y0; yf[base + t + 256] = y1; }
  if (yh) { yh[base + t] = (_Float16)y0; yh[base + t + 256] = (_Float16)y1; }
}

// ================= host orchestration =================
extern "C" void kernel_launch(void* const* d_in, const int* in_sizes, int n_in,
                              void* d_out, int out_size, void* d_ws, size_t ws_size,
                              hipStream_t stream) {
  (void)in_sizes; (void)n_in; (void)out_size; (void)ws_size;
  const float* x_in = (const float*)d_in[0];
  const float* wgw  = (const float*)d_in[21];
  const float* wgb  = (const float*)d_in[22];
  const float* btab = (const float*)d_in[23];

  char* ws = (char*)d_ws;
  size_t off = 0;
  auto alloc = [&](size_t bytes) { void* p = ws + off; off += (bytes + 255) & ~(size_t)255; return p; };
  // ~250 MB workspace plan (f16 activations + f32 residual buffers, heavy reuse)
  _Float16* XH  = (_Float16*)alloc((size_t)TOK * DD * 2);   // f16 of current x (then x1)
  _Float16* QH  = (_Float16*)alloc((size_t)TOK * DD * 2);
  _Float16* KH  = (_Float16*)alloc((size_t)TOK * DD * 2);
  _Float16* VH  = (_Float16*)alloc((size_t)TOK * DD * 2);
  _Float16* OH  = (_Float16*)alloc((size_t)TOK * DD * 2);
  _Float16* FFH = (_Float16*)alloc((size_t)TOK * DFD * 2);
  _Float16* WT0 = (_Float16*)alloc((size_t)2 * 1024 * 1024 * 2);  // transposed f16 weights scratch
  float* XA  = (float*)alloc((size_t)TOK * DD * 4);  // x1
  float* XB  = (float*)alloc((size_t)TOK * DD * 4);  // layer-0 output
  float* PRJ = (float*)alloc((size_t)TOK * DD * 4);  // o-proj / ffn output
  float* ATD = (float*)alloc((size_t)TOK * DD * 4);
  float* ATP = (float*)alloc((size_t)TOK * DD * 4);
  float* DREL = (float*)alloc((size_t)HD * SD * SD * 4);
  float* PREL = (float*)alloc((size_t)SD * SD * 4);
  int*   MSK  = (int*)alloc((size_t)BD * SD * 4);

  // batch-independent setup
  k_mask<<<BD * SD, 128, 0, stream>>>(x_in, MSK);
  k_disrel<<<(HD * SD * SD + 255) / 256, 256, 0, stream>>>(wgw, wgb, DREL);
  k_posrel<<<(SD * SD + 255) / 256, 256, 0, stream>>>(btab, PREL);
  k_f32_to_f16<<<(TOK * DD + 255) / 256, 256, 0, stream>>>(x_in, XH, TOK * DD);

  auto gemm = [&](const _Float16* A, const _Float16* BT, const float* bias,
                  float* Cf, _Float16* Ch, int M, int N, int K, int relu) {
    dim3 g(M / 128, N / 64);
    k_gemm<<<g, 512, 0, stream>>>(A, BT, bias, Cf, Ch, M, N, K, relu);
  };
  auto trans = [&](const float* w, _Float16* wt, int K, int N) {
    k_wt_transpose<<<(K * N + 255) / 256, 256, 0, stream>>>(w, wt, K, N);
  };

  const float* XCUR = x_in;
  for (int i = 0; i < NLAY; ++i) {
    float* X2 = (i == 0) ? XB : (float*)d_out;

    for (int t = 0; t < 2; ++t) {  // t=0: dis branch, t=1: pos branch
      int bs = 1 + t * 10;
      const float* wq = (const float*)d_in[bs + 0] + (size_t)i * DD * DD;
      const float* bq = (const float*)d_in[bs + 1] + (size_t)i * DD;
      const float* wk = (const float*)d_in[bs + 2] + (size_t)i * DD * DD;
      const float* bk = (const float*)d_in[bs + 3] + (size_t)i * DD;
      const float* wv = (const float*)d_in[bs + 4] + (size_t)i * DD * DD;
      const float* bv = (const float*)d_in[bs + 5] + (size_t)i * DD;
      const float* wo = (const float*)d_in[bs + 6] + (size_t)i * DD * DD;
      const float* bo = (const float*)d_in[bs + 7] + (size_t)i * DD;
      const float* lg = (const float*)d_in[bs + 8] + (size_t)i * DD;
      const float* lb = (const float*)d_in[bs + 9] + (size_t)i * DD;
      _Float16* WQ = WT0;
      _Float16* WK = WT0 + 262144;
      _Float16* WV = WT0 + 524288;
      _Float16* WO = WT0 + 786432;
      trans(wq, WQ, DD, DD); trans(wk, WK, DD, DD);
      trans(wv, WV, DD, DD); trans(wo, WO, DD, DD);
      gemm(XH, WQ, bq, nullptr, QH, TOK, DD, DD, 0);
      gemm(XH, WK, bk, nullptr, KH, TOK, DD, DD, 0);
      gemm(XH, WV, bv, nullptr, VH, TOK, DD, DD, 0);
      k_attn<<<BD * HD, 256, 0, stream>>>(QH, KH, VH,
          (t == 0) ? DREL : PREL, (t == 0) ? 1 : 0, MSK, OH);
      gemm(OH, WO, bo, PRJ, nullptr, TOK, DD, DD, 0);
      k_res_ln<<<TOK, 256, 0, stream>>>(XCUR, PRJ, nullptr, 1.f, 0.f, lg, lb,
                                        (t == 0) ? ATD : ATP, nullptr);
    }

    // x1 = LN(x + 0.5*att_d + 0.5*att_p)   (f32 -> XA, f16 -> XH for FFN)
    k_res_ln<<<TOK, 256, 0, stream>>>(XCUR, ATD, ATP, 0.5f, 0.5f,
        (const float*)d_in[24] + (size_t)i * DD, (const float*)d_in[25] + (size_t)i * DD,
        XA, XH);

    // FFN
    trans((const float*)d_in[26] + (size_t)i * DD * DFD, WT0, DD, DFD);
    trans((const float*)d_in[28] + (size_t)i * DFD * DD, WT0 + 1048576, DFD, DD);
    gemm(XH, WT0, (const float*)d_in[27] + (size_t)i * DFD, nullptr, FFH, TOK, DFD, DD, 1);
    gemm(FFH, WT0 + 1048576, (const float*)d_in[29] + (size_t)i * DD, PRJ, nullptr, TOK, DD, DFD, 0);

    // x2 = LN(x1 + ff); last layer writes d_out, others also refresh f16 x for next layer
    k_res_ln<<<TOK, 256, 0, stream>>>(XA, PRJ, nullptr, 1.f, 0.f,
        (const float*)d_in[30] + (size_t)i * DD, (const float*)d_in[31] + (size_t)i * DD,
        X2, (i < NLAY - 1) ? XH : nullptr);
    XCUR = X2;
  }
}
